// RayTransformer_43619687858633
// MI455X (gfx1250) — matrix operand order repaired and checked by reading, not compiled, not run on hardware
//
#include <hip/hip_runtime.h>
#include <math.h>

typedef __attribute__((ext_vector_type(16))) _Float16 v16h;
typedef __attribute__((ext_vector_type(8)))  float    v8f;

#define B_   2
#define NV_  4
#define RN_  1024
#define SN_  128
#define CF_  32
#define H_   512
#define W_   640
#define HF_  128
#define WF_  160
#define P_   (B_*RN_*SN_)   /* 262144 points */

#define WMMA_F16(a,b,c) __builtin_amdgcn_wmma_f32_16x16x32_f16(false,(a),false,(b),(short)0,(c),false,false)

static __device__ __forceinline__ v8f z8() {
  v8f z = {0.f,0.f,0.f,0.f,0.f,0.f,0.f,0.f};
  return z;
}
static __device__ __forceinline__ int imin_(int a,int b){return a<b?a:b;}
static __device__ __forceinline__ int imax_(int a,int b){return a>b?a:b;}

// ---- WMMA fragment loaders (wave32, 16x16x32 f16) ----
// A: lane<16 -> row M0+lane, K = kb + {0..7, 16..23}; lane>=16 -> row M0+lane-16, K+8
static __device__ __forceinline__ v16h ld_a(const _Float16* s, int stride, int m0, int kb) {
  int lane = threadIdx.x & 31;
  const _Float16* p = s + (size_t)(m0 + (lane & 15)) * stride + kb + ((lane & 16) ? 8 : 0);
  v16h a;
#pragma unroll
  for (int e = 0; e < 8; ++e) { a[e] = p[e]; a[e + 8] = p[e + 16]; }
  return a;
}
// B (stored transposed: Bt[n][k], K contiguous): lane<16 -> col n0+lane, K = kb + {0..7, 16..23}; lane>=16 -> K+8
static __device__ __forceinline__ v16h ld_b(const _Float16* s, int stride, int n0, int kb) {
  int lane = threadIdx.x & 31;
  const _Float16* p = s + (size_t)(n0 + (lane & 15)) * stride + kb + ((lane & 16) ? 8 : 0);
  v16h b;
#pragma unroll
  for (int e = 0; e < 8; ++e) { b[e] = p[e]; b[e + 8] = p[e + 16]; }
  return b;
}
// D/C: VGPR r -> row m0 + r + (lane>=16 ? 8:0), col n0 + (lane&15)
// act: 0=none, 1=elu+1, 2=relu
static __device__ __forceinline__ void st_f16(v8f d, _Float16* out, int stride,
                                              int m0, int n0, int ncap, int act, float scale) {
  int lane = threadIdx.x & 31;
  int col = n0 + (lane & 15);
  int rb  = m0 + ((lane & 16) ? 8 : 0);
  if (col < ncap) {
#pragma unroll
    for (int r = 0; r < 8; ++r) {
      float v = d[r];
      if (act == 1)      v = (v > 0.f) ? (v + 1.f) : __expf(v);
      else if (act == 2) v = fmaxf(v, 0.f);
      v *= scale;
      out[(size_t)(rb + r) * stride + col] = (_Float16)v;
    }
  }
}
static __device__ __forceinline__ void st_f32(v8f d, float* out, int stride,
                                              int m0, int n0, int ncap, int act, float scale) {
  int lane = threadIdx.x & 31;
  int col = n0 + (lane & 15);
  int rb  = m0 + ((lane & 16) ? 8 : 0);
  if (col < ncap) {
#pragma unroll
    for (int r = 0; r < 8; ++r) {
      float v = d[r];
      if (act == 1)      v = (v > 0.f) ? (v + 1.f) : __expf(v);
      else if (act == 2) v = fmaxf(v, 0.f);
      out[(size_t)(rb + r) * stride + col] = v * scale;
    }
  }
}

// ============================================================================
// Kernel 0: positional encoding table (SN x 8)
// ============================================================================
__global__ void pe_kernel(float* __restrict__ pe) {
  int t = blockIdx.x * blockDim.x + threadIdx.x;
  if (t < SN_ * 8) {
    int s = t >> 3, j = t & 7;
    float ex = (float)(2 * (j / 2)) / 8.0f;
    float ang = (float)s / __powf(10000.0f, ex);
    pe[t] = (j & 1) ? __cosf(ang) : __sinf(ang);
  }
}

// ============================================================================
// Kernel 1: geometry + projection + bilinear sampling
// one thread per (b, nv, r, s)
// ============================================================================
__global__ __launch_bounds__(256) void geom_kernel(
    const float* __restrict__ point3D, const float* __restrict__ simgs,
    const float* __restrict__ poses,   const float* __restrict__ poses_inv,
    const float* __restrict__ refpi,   const float* __restrict__ feat,
    const float* __restrict__ vtok,
    float* __restrict__ xview, float* __restrict__ dirb, float* __restrict__ maskb,
    float* __restrict__ rgbb,  float* __restrict__ outPip)
{
  int idx = blockIdx.x * 256 + threadIdx.x;
  if (idx >= B_ * NV_ * RN_ * SN_) return;
  int s  = idx & (SN_ - 1);
  int r  = (idx >> 7)  & (RN_ - 1);
  int n  = (idx >> 17) & 3;
  int bb = idx >> 19;
  int gp = (bb * RN_ + r) * SN_ + s;

  float px = point3D[(size_t)gp * 3 + 0];
  float py = point3D[(size_t)gp * 3 + 1];
  float pz = point3D[(size_t)gp * 3 + 2];

  const float* rp = refpi + bb * 16;
  float v1x = px - rp[3], v1y = py - rp[7], v1z = pz - rp[11];
  float il = rsqrtf(v1x * v1x + v1y * v1y + v1z * v1z);
  v1x *= il; v1y *= il; v1z *= il;

  const float* ip = poses_inv + (bb * NV_ + n) * 16;
  float v2x = px - ip[3], v2y = py - ip[7], v2z = pz - ip[11];
  float il2 = rsqrtf(v2x * v2x + v2y * v2y + v2z * v2z);
  v2x *= il2; v2y *= il2; v2z *= il2;

  size_t db = ((size_t)gp * NV_ + n) * 3;
  dirb[db + 0] = v1x - v2x; dirb[db + 1] = v1y - v2y; dirb[db + 2] = v1z - v2z;

  const float* Pm = poses + (bb * NV_ + n) * 16;
  float pr0 = Pm[0] * px + Pm[1] * py + Pm[2]  * pz + Pm[3];
  float pr1 = Pm[4] * px + Pm[5] * py + Pm[6]  * pz + Pm[7];
  float pr2 = Pm[8] * px + Pm[9] * py + Pm[10] * pz + Pm[11];
  float gx = pr0 / pr2, gy = pr1 / pr2;

  size_t pipb = ((size_t)((bb * NV_ + n) * 2) * RN_ + r) * SN_ + s;
  outPip[pipb] = gx;
  outPip[pipb + (size_t)RN_ * SN_] = gy;

  float mdep = (pr2 > 0.f) ? 1.f : 0.f;
  float inb  = (gx >= 0.f && gx <= (float)(WF_ - 1) && gy >= 0.f && gy <= (float)(HF_ - 1)) ? 1.f : 0.f;
  maskb[(size_t)gp * NV_ + n] = inb * mdep;

  float fxf = floorf(gx), fyf = floorf(gy);
  int ix0 = (int)fxf, iy0 = (int)fyf;
  float wx = gx - fxf, wy = gy - fyf;
  int   cxs[4] = {ix0, ix0 + 1, ix0, ix0 + 1};
  int   cys[4] = {iy0, iy0, iy0 + 1, iy0 + 1};
  float cw[4]  = {(1.f - wx) * (1.f - wy), wx * (1.f - wy), (1.f - wx) * wy, wx * wy};

  // ---- feature sampling (HF x WF, CF channels) ----
  {
    int lin[4]; float w4[4];
#pragma unroll
    for (int j = 0; j < 4; ++j) {
      bool ok = (cxs[j] >= 0 && cxs[j] < WF_ && cys[j] >= 0 && cys[j] < HF_);
      int xx = imin_(imax_(cxs[j], 0), WF_ - 1);
      int yy = imin_(imax_(cys[j], 0), HF_ - 1);
      lin[j] = yy * WF_ + xx;
      w4[j] = ok ? cw[j] : 0.f;
    }
    const float* fb = feat + (size_t)(bb * NV_ + n) * (CF_ * HF_ * WF_);
    __builtin_prefetch(fb + lin[0], 0, 1);
    float acc[CF_];
#pragma unroll
    for (int c = 0; c < CF_; ++c) acc[c] = 0.f;
#pragma unroll
    for (int j = 0; j < 4; ++j) {
      float w = w4[j];
      if (w != 0.f) {
        const float* fp = fb + lin[j];
#pragma unroll
        for (int c = 0; c < CF_; ++c) acc[c] += w * fp[(size_t)c * (HF_ * WF_)];
      }
    }
    size_t xb = (size_t)gp * 160 + (size_t)(n + 1) * 32;
#pragma unroll
    for (int c = 0; c < CF_; ++c) xview[xb + c] = acc[c];
    if (n == 0) {
#pragma unroll
      for (int c = 0; c < CF_; ++c) xview[(size_t)gp * 160 + c] = vtok[c];
    }
  }
  // ---- rgb sampling (H x W, 3 channels) ----
  {
    int lin[4]; float w4[4];
#pragma unroll
    for (int j = 0; j < 4; ++j) {
      bool ok = (cxs[j] >= 0 && cxs[j] < W_ && cys[j] >= 0 && cys[j] < H_);
      int xx = imin_(imax_(cxs[j], 0), W_ - 1);
      int yy = imin_(imax_(cys[j], 0), H_ - 1);
      lin[j] = yy * W_ + xx;
      w4[j] = ok ? cw[j] : 0.f;
    }
    const float* ib = simgs + (size_t)(bb * NV_ + n) * 3 * H_ * W_;
    float acc[3] = {0.f, 0.f, 0.f};
#pragma unroll
    for (int j = 0; j < 4; ++j) {
      float w = w4[j];
      if (w != 0.f) {
        const float* fp = ib + lin[j];
#pragma unroll
        for (int c = 0; c < 3; ++c) acc[c] += w * fp[(size_t)c * (H_ * W_)];
      }
    }
    size_t rb = (size_t)gp * 12 + (size_t)n * 3;
#pragma unroll
    for (int c = 0; c < 3; ++c) rgbb[rb + c] = acc[c];
  }
}

// ============================================================================
// Kernel 2: fused view transformer layer (L=5, d=32) — LDS resident, WMMA.
// 16 points per 128-thread workgroup => 80 rows. Updates xview in place.
// ============================================================================
__global__ __launch_bounds__(128) void view_layer_kernel(
    float* __restrict__ xview,
    const float* __restrict__ Wq, const float* __restrict__ Wk,
    const float* __restrict__ Wv, const float* __restrict__ Wm,
    const float* __restrict__ W1, const float* __restrict__ W2,
    const float* __restrict__ ln1w, const float* __restrict__ ln1b,
    const float* __restrict__ ln2w, const float* __restrict__ ln2b)
{
  __shared__ __align__(16) char vsm[46592];
  _Float16* wqT  = (_Float16*)(vsm + 0);      // 32x32 (n-major, k-contig)
  _Float16* wkT  = (_Float16*)(vsm + 2048);
  _Float16* wvT  = (_Float16*)(vsm + 4096);
  _Float16* wmT  = (_Float16*)(vsm + 6144);
  _Float16* w1T  = (_Float16*)(vsm + 8192);   // 64x64
  _Float16* w2T  = (_Float16*)(vsm + 16384);  // 32x64
  float*    lnP  = (float*)(vsm + 20480);     // 4x32
  _Float16* sX   = (_Float16*)(vsm + 20992);  // 80x32
  _Float16* sQ   = (_Float16*)(vsm + 26112);  // 80x32
  _Float16* sK   = (_Float16*)(vsm + 31232);
  _Float16* sV   = (_Float16*)(vsm + 36352);
  _Float16* sMsg = (_Float16*)(vsm + 41472);
  _Float16* sMerge = sQ;        // merge output over (dead) q
  _Float16* sH1    = sQ;        // 80x64 over sQ+sK
  float*    sH2    = (float*)sV;// 80x32 f32 over sV+sMsg

  int tid = threadIdx.x, wave = tid >> 5;
  int gp0 = blockIdx.x * 16;

  for (int i = tid; i < 1024; i += 128) {
    int n = i >> 5, k = i & 31;
    wqT[i] = (_Float16)Wq[k * 32 + n];
    wkT[i] = (_Float16)Wk[k * 32 + n];
    wvT[i] = (_Float16)Wv[k * 32 + n];
    wmT[i] = (_Float16)Wm[k * 32 + n];
  }
  for (int i = tid; i < 4096; i += 128) { int n = i >> 6, k = i & 63; w1T[i] = (_Float16)W1[k * 64 + n]; }
  for (int i = tid; i < 2048; i += 128) { int n = i >> 6, k = i & 63; w2T[i] = (_Float16)W2[k * 32 + n]; }
  for (int i = tid; i < 32; i += 128) {
    lnP[i] = ln1w[i]; lnP[32 + i] = ln1b[i]; lnP[64 + i] = ln2w[i]; lnP[96 + i] = ln2b[i];
  }
  for (int i = tid; i < 2560; i += 128) {
    int row = i >> 5, c = i & 31;
    int pb = row / 5, tok = row - pb * 5;
    sX[i] = (_Float16)xview[(size_t)(gp0 + pb) * 160 + tok * 32 + c];
  }
  __syncthreads();

  // q/k/v projections (M=80, N=32, K=32): 10 tiles, A reused across 3 GEMMs
  for (int t = wave; t < 10; t += 4) {
    int m0 = (t % 5) * 16, n0 = (t / 5) * 16;
    v16h a = ld_a(sX, 32, m0, 0);
    v8f aq = z8(), ak = z8(), av = z8();
    aq = WMMA_F16(a, ld_b(wqT, 32, n0, 0), aq);
    ak = WMMA_F16(a, ld_b(wkT, 32, n0, 0), ak);
    av = WMMA_F16(a, ld_b(wvT, 32, n0, 0), av);
    st_f16(aq, sQ, 32, m0, n0, 32, 1, 1.f);        // elu+1
    st_f16(ak, sK, 32, m0, n0, 32, 1, 1.f);        // elu+1
    st_f16(av, sV, 32, m0, n0, 32, 0, 1.f / 5.f);  // v / L
  }
  __syncthreads();

  // linear attention per (point, head): 16 points x 8 heads = 128 threads
  {
    int pb = tid >> 3, h = tid & 7;
    float kv[4][4] = {{0.f}};
    float ks[4] = {0.f, 0.f, 0.f, 0.f};
    for (int s = 0; s < 5; ++s) {
      int row = pb * 5 + s;
      float kk[4], vv[4];
#pragma unroll
      for (int d = 0; d < 4; ++d) {
        kk[d] = (float)sK[row * 32 + h * 4 + d];
        vv[d] = (float)sV[row * 32 + h * 4 + d];
      }
#pragma unroll
      for (int d = 0; d < 4; ++d) {
        ks[d] += kk[d];
#pragma unroll
        for (int u = 0; u < 4; ++u) kv[d][u] += kk[d] * vv[u];
      }
    }
    for (int s = 0; s < 5; ++s) {
      int row = pb * 5 + s;
      float qq[4];
#pragma unroll
      for (int d = 0; d < 4; ++d) qq[d] = (float)sQ[row * 32 + h * 4 + d];
      float den = 1e-6f;
#pragma unroll
      for (int d = 0; d < 4; ++d) den += qq[d] * ks[d];
      float zz = 1.f / den;
#pragma unroll
      for (int u = 0; u < 4; ++u) {
        float acc = 0.f;
#pragma unroll
        for (int d = 0; d < 4; ++d) acc += qq[d] * kv[d][u];
        sMsg[row * 32 + h * 4 + u] = (_Float16)(5.f * zz * acc);
      }
    }
  }
  __syncthreads();

  // merge GEMM -> sMerge (over dead q)
  for (int t = wave; t < 10; t += 4) {
    int m0 = (t % 5) * 16, n0 = (t / 5) * 16;
    v8f acc = z8();
    acc = WMMA_F16(ld_a(sMsg, 32, m0, 0), ld_b(wmT, 32, n0, 0), acc);
    st_f16(acc, sMerge, 32, m0, n0, 32, 0, 1.f);
  }
  __syncthreads();

  // LayerNorm1 (d=32): msgln -> sMsg
  if (tid < 80) {
    float mu = 0.f;
    for (int c = 0; c < 32; ++c) mu += (float)sMerge[tid * 32 + c];
    mu *= (1.f / 32.f);
    float var = 0.f;
    for (int c = 0; c < 32; ++c) { float d = (float)sMerge[tid * 32 + c] - mu; var += d * d; }
    var *= (1.f / 32.f);
    float rs = rsqrtf(var + 1e-5f);
    for (int c = 0; c < 32; ++c)
      sMsg[tid * 32 + c] = (_Float16)(((float)sMerge[tid * 32 + c] - mu) * rs * lnP[c] + lnP[32 + c]);
  }
  __syncthreads();

  // mlp1: concat(x, msgln) (K=64) -> 64, relu
  for (int t = wave; t < 20; t += 4) {
    int m0 = (t % 5) * 16, n0 = (t / 5) * 16;
    v8f acc = z8();
    acc = WMMA_F16(ld_a(sX,   32, m0, 0), ld_b(w1T, 64, n0, 0),  acc);
    acc = WMMA_F16(ld_a(sMsg, 32, m0, 0), ld_b(w1T, 64, n0, 32), acc);
    st_f16(acc, sH1, 64, m0, n0, 64, 2, 1.f);
  }
  __syncthreads();

  // mlp2: 64 -> 32
  for (int t = wave; t < 10; t += 4) {
    int m0 = (t % 5) * 16, n0 = (t / 5) * 16;
    v8f acc = z8();
    acc = WMMA_F16(ld_a(sH1, 64, m0, 0),  ld_b(w2T, 64, n0, 0),  acc);
    acc = WMMA_F16(ld_a(sH1, 64, m0, 32), ld_b(w2T, 64, n0, 32), acc);
    st_f32(acc, sH2, 32, m0, n0, 32, 0, 1.f);
  }
  __syncthreads();

  // residual: x += LN2(h)
  if (tid < 80) {
    int pb = tid / 5, tok = tid - pb * 5;
    float mu = 0.f;
    for (int c = 0; c < 32; ++c) mu += sH2[tid * 32 + c];
    mu *= (1.f / 32.f);
    float var = 0.f;
    for (int c = 0; c < 32; ++c) { float d = sH2[tid * 32 + c] - mu; var += d * d; }
    var *= (1.f / 32.f);
    float rs = rsqrtf(var + 1e-5f);
    size_t base = (size_t)(gp0 + pb) * 160 + tok * 32;
    for (int c = 0; c < 32; ++c)
      xview[base + c] = xview[base + c] + ((sH2[tid * 32 + c] - mu) * rs * lnP[64 + c] + lnP[96 + c]);
  }
}

// ============================================================================
// Kernel 3: fused ray transformer layer (L=128, d=40) — one ray per workgroup,
// fully LDS resident, WMMA GEMMs (K padded to 64/96). Writes xr_out (P,40).
// ============================================================================
#define RAY_SMEM 142912

__global__ __launch_bounds__(256) void ray_layer_kernel(
    const float* __restrict__ xview, const float* __restrict__ pe,
    const float* __restrict__ Wq, const float* __restrict__ Wk,
    const float* __restrict__ Wv, const float* __restrict__ Wm,
    const float* __restrict__ W1, const float* __restrict__ W2,
    const float* __restrict__ ln1w, const float* __restrict__ ln1b,
    const float* __restrict__ ln2w, const float* __restrict__ ln2b,
    float* __restrict__ xrout)
{
  extern __shared__ __align__(16) char rsm[];
  _Float16* wqT  = (_Float16*)(rsm + 0);       // 48x64 (zero padded)
  _Float16* wkT  = (_Float16*)(rsm + 6144);
  _Float16* wvT  = (_Float16*)(rsm + 12288);
  _Float16* wmT  = (_Float16*)(rsm + 18432);
  _Float16* w1T  = (_Float16*)(rsm + 24576);   // 80x96
  _Float16* w2T  = (_Float16*)(rsm + 39936);   // 48x96
  float*    lnP  = (float*)(rsm + 49152);      // 4x40
  _Float16* sX   = (_Float16*)(rsm + 49792);   // 128x64 (cols 40..63 = 0)
  _Float16* sQ   = (_Float16*)(rsm + 66176);   // 128x40
  _Float16* sK   = (_Float16*)(rsm + 76416);
  _Float16* sV   = (_Float16*)(rsm + 86656);
  _Float16* sMsg = (_Float16*)(rsm + 96896);   // 128x64
  _Float16* sCat = (_Float16*)(rsm + 113280);  // 128x96
  float*    sKV  = (float*)(rsm + 137856);     // 8*5*5
  float*    sKs  = (float*)(rsm + 138656);     // 40
  float*    sZ   = (float*)(rsm + 138816);     // 128x8
  _Float16* sMerge = sQ;          // after attention, q/k/v dead
  _Float16* sH1    = sQ;          // 128x96 spans sQ..sV
  float*    sH2    = (float*)sCat;// 128x40 f32

  int tid = threadIdx.x, wave = tid >> 5;
  int gp0 = blockIdx.x * SN_;

  for (int i = tid; i < 48 * 64; i += 256) {
    int n = i >> 6, k = i & 63;
    _Float16 q_ = (_Float16)0.f, k_ = (_Float16)0.f, v_ = (_Float16)0.f, m_ = (_Float16)0.f;
    if (n < 40 && k < 40) {
      q_ = (_Float16)Wq[k * 40 + n]; k_ = (_Float16)Wk[k * 40 + n];
      v_ = (_Float16)Wv[k * 40 + n]; m_ = (_Float16)Wm[k * 40 + n];
    }
    wqT[i] = q_; wkT[i] = k_; wvT[i] = v_; wmT[i] = m_;
  }
  for (int i = tid; i < 80 * 96; i += 256) {
    int n = i / 96, k = i % 96;
    w1T[i] = (k < 80) ? (_Float16)W1[k * 80 + n] : (_Float16)0.f;
  }
  for (int i = tid; i < 48 * 96; i += 256) {
    int n = i / 96, k = i % 96;
    w2T[i] = (n < 40 && k < 80) ? (_Float16)W2[k * 40 + n] : (_Float16)0.f;
  }
  for (int i = tid; i < 40; i += 256) {
    lnP[i] = ln1w[i]; lnP[40 + i] = ln1b[i]; lnP[80 + i] = ln2w[i]; lnP[120 + i] = ln2b[i];
  }
  // xr = [x0(32) | posenc(8) | zeros(24)]
  for (int i = tid; i < 128 * 64; i += 256) {
    int row = i >> 6, c = i & 63;
    float v = 0.f;
    if (c < 32)      v = xview[(size_t)(gp0 + row) * 160 + c];
    else if (c < 40) v = pe[row * 8 + (c - 32)];
    sX[i] = (_Float16)v;
  }
  __syncthreads();

  // q/k/v: M=128 (8 tiles), N=40 (3 tiles), K=40 (2 k-steps, zero padded)
  for (int t = wave; t < 24; t += 8) {
    int m0 = (t & 7) * 16, n0 = (t >> 3) * 16;
    v8f aq = z8(), ak = z8(), av = z8();
#pragma unroll
    for (int ks = 0; ks < 2; ++ks) {
      v16h a = ld_a(sX, 64, m0, ks * 32);
      aq = WMMA_F16(a, ld_b(wqT, 64, n0, ks * 32), aq);
      ak = WMMA_F16(a, ld_b(wkT, 64, n0, ks * 32), ak);
      av = WMMA_F16(a, ld_b(wvT, 64, n0, ks * 32), av);
    }
    st_f16(aq, sQ, 40, m0, n0, 40, 1, 1.f);
    st_f16(ak, sK, 40, m0, n0, 40, 1, 1.f);
    st_f16(av, sV, 40, m0, n0, 40, 0, 1.f / 128.f);
  }
  __syncthreads();

  // kv (8 heads x 5x5) and ksum(40)
  if (tid < 200) {
    int h = tid / 25, d = (tid / 5) % 5, u = tid % 5;
    float acc = 0.f;
    for (int s = 0; s < 128; ++s)
      acc += (float)sK[s * 40 + h * 5 + d] * (float)sV[s * 40 + h * 5 + u];
    sKV[tid] = acc;
  } else if (tid < 240) {
    int c = tid - 200;
    float acc = 0.f;
    for (int s = 0; s < 128; ++s) acc += (float)sK[s * 40 + c];
    sKs[c] = acc;
  }
  __syncthreads();

  // z[s][h] = 1/(q . ksum + eps)
  for (int i = tid; i < 128 * 8; i += 256) {
    int s = i >> 3, h = i & 7;
    float den = 1e-6f;
#pragma unroll
    for (int d = 0; d < 5; ++d) den += (float)sQ[s * 40 + h * 5 + d] * sKs[h * 5 + d];
    sZ[i] = 1.f / den;
  }
  __syncthreads();

  // msg[s][c] = L * z * (q . kv)
  for (int i = tid; i < 128 * 64; i += 256) {
    int s = i >> 6, c = i & 63;
    float v = 0.f;
    if (c < 40) {
      int h = c / 5, u = c % 5;
      float acc = 0.f;
#pragma unroll
      for (int d = 0; d < 5; ++d) acc += (float)sQ[s * 40 + h * 5 + d] * sKV[(h * 5 + d) * 5 + u];
      v = 128.f * sZ[s * 8 + h] * acc;
    }
    sMsg[i] = (_Float16)v;
  }
  __syncthreads();

  // merge GEMM -> sMerge (over dead q)
  for (int t = wave; t < 24; t += 8) {
    int m0 = (t & 7) * 16, n0 = (t >> 3) * 16;
    v8f acc = z8();
#pragma unroll
    for (int ks = 0; ks < 2; ++ks)
      acc = WMMA_F16(ld_a(sMsg, 64, m0, ks * 32), ld_b(wmT, 64, n0, ks * 32), acc);
    st_f16(acc, sMerge, 40, m0, n0, 40, 0, 1.f);
  }
  __syncthreads();

  // LN1 + build concat [xr | msgln | 0] (128x96)
  if (tid < 128) {
    float mu = 0.f;
    for (int c = 0; c < 40; ++c) mu += (float)sMerge[tid * 40 + c];
    mu *= (1.f / 40.f);
    float var = 0.f;
    for (int c = 0; c < 40; ++c) { float d = (float)sMerge[tid * 40 + c] - mu; var += d * d; }
    var *= (1.f / 40.f);
    float rs = rsqrtf(var + 1e-5f);
    for (int c = 0; c < 96; ++c) {
      _Float16 v;
      if (c < 40)      v = sX[tid * 64 + c];
      else if (c < 80) {
        float t0 = (float)sMerge[tid * 40 + (c - 40)];
        v = (_Float16)((t0 - mu) * rs * lnP[c - 40] + lnP[40 + (c - 40)]);
      } else v = (_Float16)0.f;
      sCat[tid * 96 + c] = v;
    }
  }
  __syncthreads();

  // zero-fill sH1 padding cols (80..95), then mlp1 (relu)
  for (int i = tid; i < 128 * 16; i += 256) {
    int row = i >> 4, c = 80 + (i & 15);
    sH1[row * 96 + c] = (_Float16)0.f;
  }
  for (int t = wave; t < 40; t += 8) {
    int m0 = (t & 7) * 16, n0 = (t >> 3) * 16;
    v8f acc = z8();
#pragma unroll
    for (int ks = 0; ks < 3; ++ks)
      acc = WMMA_F16(ld_a(sCat, 96, m0, ks * 32), ld_b(w1T, 96, n0, ks * 32), acc);
    st_f16(acc, sH1, 96, m0, n0, 80, 2, 1.f);
  }
  __syncthreads();

  // mlp2 -> sH2 (f32, over dead sCat)
  for (int t = wave; t < 24; t += 8) {
    int m0 = (t & 7) * 16, n0 = (t >> 3) * 16;
    v8f acc = z8();
#pragma unroll
    for (int ks = 0; ks < 3; ++ks)
      acc = WMMA_F16(ld_a(sH1, 96, m0, ks * 32), ld_b(w2T, 96, n0, ks * 32), acc);
    st_f32(acc, sH2, 40, m0, n0, 40, 0, 1.f);
  }
  __syncthreads();

  // residual: xr_out = xr + LN2(h)
  if (tid < 128) {
    float mu = 0.f;
    for (int c = 0; c < 40; ++c) mu += sH2[tid * 40 + c];
    mu *= (1.f / 40.f);
    float var = 0.f;
    for (int c = 0; c < 40; ++c) { float d = sH2[tid * 40 + c] - mu; var += d * d; }
    var *= (1.f / 40.f);
    float rs = rsqrtf(var + 1e-5f);
    size_t ob = (size_t)(gp0 + tid) * 40;
    for (int c = 0; c < 40; ++c)
      xrout[ob + c] = (float)sX[tid * 64 + c] + ((sH2[tid * 40 + c] - mu) * rs * lnP[80 + c] + lnP[120 + c]);
  }
}

// ============================================================================
// Kernel 4: density MLP 40->32->16->1 (weights in LDS, one point per thread)
// ============================================================================
__global__ __launch_bounds__(256) void density_kernel(
    const float* __restrict__ xr,
    const float* __restrict__ w1, const float* __restrict__ b1,
    const float* __restrict__ w2, const float* __restrict__ b2,
    const float* __restrict__ w3, const float* __restrict__ b3,
    float* __restrict__ out)
{
  __shared__ float sw1[40 * 32], sb1[32], sw2[32 * 16], sb2[16], sw3[16], sb3v[1];
  int tid = threadIdx.x;
  for (int i = tid; i < 40 * 32; i += 256) sw1[i] = w1[i];
  for (int i = tid; i < 32 * 16; i += 256) sw2[i] = w2[i];
  if (tid < 32) sb1[tid] = b1[tid];
  if (tid < 16) { sb2[tid] = b2[tid]; sw3[tid] = w3[tid]; }
  if (tid == 0) sb3v[0] = b3[0];
  __syncthreads();
  int gp = blockIdx.x * 256 + tid;
  if (gp >= P_) return;
  float x[40];
#pragma unroll
  for (int k = 0; k < 40; ++k) x[k] = xr[(size_t)gp * 40 + k];
  float h1[32];
#pragma unroll
  for (int j = 0; j < 32; ++j) {
    float a = sb1[j];
    for (int k = 0; k < 40; ++k) a += x[k] * sw1[k * 32 + j];
    h1[j] = fmaxf(a, 0.f);
  }
  float h2[16];
#pragma unroll
  for (int j = 0; j < 16; ++j) {
    float a = sb2[j];
    for (int k = 0; k < 32; ++k) a += h1[k] * sw2[k * 16 + j];
    h2[j] = fmaxf(a, 0.f);
  }
  float o = sb3v[0];
#pragma unroll
  for (int k = 0; k < 16; ++k) o += h2[k] * sw3[k];
  out[gp] = o;
}

// ============================================================================
// Kernel 5: radiance MLP 35->16->8->1 per (point, view) + masked softmax blend
// ============================================================================
__global__ __launch_bounds__(256) void radiance_kernel(
    const float* __restrict__ xview, const float* __restrict__ dirb,
    const float* __restrict__ maskb, const float* __restrict__ rgbb,
    const float* __restrict__ w1, const float* __restrict__ b1,
    const float* __restrict__ w2, const float* __restrict__ b2,
    const float* __restrict__ w3, const float* __restrict__ b3,
    float* __restrict__ outRad)
{
  __shared__ float cw1[35 * 16], cb1[16], cw2[16 * 8], cb2[8], cw3[8], cb3v[1];
  int tid = threadIdx.x;
  for (int i = tid; i < 35 * 16; i += 256) cw1[i] = w1[i];
  for (int i = tid; i < 16 * 8; i += 256) cw2[i] = w2[i];
  if (tid < 16) cb1[tid] = b1[tid];
  if (tid < 8)  { cb2[tid] = b2[tid]; cw3[tid] = w3[tid]; }
  if (tid == 0) cb3v[0] = b3[0];
  __syncthreads();
  int gp = blockIdx.x * 256 + tid;
  if (gp >= P_) return;
  float xw[NV_];
#pragma unroll
  for (int v = 0; v < NV_; ++v) {
    float xin[35];
#pragma unroll
    for (int c = 0; c < 32; ++c) xin[c] = xview[(size_t)gp * 160 + (size_t)(v + 1) * 32 + c];
#pragma unroll
    for (int j = 0; j < 3; ++j) xin[32 + j] = dirb[((size_t)gp * NV_ + v) * 3 + j];
    float h1[16];
#pragma unroll
    for (int j = 0; j < 16; ++j) {
      float a = cb1[j];
      for (int k = 0; k < 35; ++k) a += xin[k] * cw1[k * 16 + j];
      h1[j] = fmaxf(a, 0.f);
    }
    float h2[8];
#pragma unroll
    for (int j = 0; j < 8; ++j) {
      float a = cb2[j];
      for (int k = 0; k < 16; ++k) a += h1[k] * cw2[k * 8 + j];
      h2[j] = fmaxf(a, 0.f);
    }
    float o = cb3v[0];
#pragma unroll
    for (int k = 0; k < 8; ++k) o += h2[k] * cw3[k];
    float m = maskb[(size_t)gp * NV_ + v];
    xw[v] = (m == 0.f) ? -1e9f : o;
  }
  float mx = fmaxf(fmaxf(xw[0], xw[1]), fmaxf(xw[2], xw[3]));
  float e[NV_], ssum = 0.f;
#pragma unroll
  for (int v = 0; v < NV_; ++v) { e[v] = __expf(xw[v] - mx); ssum += e[v]; }
  float inv = 1.f / ssum;
#pragma unroll
  for (int c = 0; c < 3; ++c) {
    float acc = 0.f;
#pragma unroll
    for (int v = 0; v < NV_; ++v) acc += (e[v] * inv) * rgbb[((size_t)gp * NV_ + v) * 3 + c];
    outRad[(size_t)gp * 3 + c] = acc;
  }
}

// ============================================================================
// Host launcher
// ============================================================================
extern "C" void kernel_launch(void* const* d_in, const int* in_sizes, int n_in,
                              void* d_out, int out_size, void* d_ws, size_t ws_size,
                              hipStream_t stream) {
  (void)in_sizes; (void)out_size;
  if (n_in < 39) return;

  const float* point3D   = (const float*)d_in[0];
  const float* simgs     = (const float*)d_in[1];
  const float* poses     = (const float*)d_in[2];
  const float* poses_inv = (const float*)d_in[3];
  const float* refpi     = (const float*)d_in[4];
  const float* feat      = (const float*)d_in[5];
  const float* vtok      = (const float*)d_in[6];
  const float* vW[10]; for (int i = 0; i < 10; ++i) vW[i] = (const float*)d_in[7 + i];
  const float* rW[10]; for (int i = 0; i < 10; ++i) rW[i] = (const float*)d_in[17 + i];
  const float* dW[6];  for (int i = 0; i < 6;  ++i) dW[i] = (const float*)d_in[27 + i];
  const float* cW[6];  for (int i = 0; i < 6;  ++i) cW[i] = (const float*)d_in[33 + i];

  // workspace layout (floats)
  float* ws    = (float*)d_ws;
  float* xview = ws;                              // P x 5 x 32
  float* dirb  = xview + (size_t)P_ * 160;        // P x 4 x 3
  float* maskb = dirb  + (size_t)P_ * 12;         // P x 4
  float* rgbb  = maskb + (size_t)P_ * 4;          // P x 4 x 3
  float* xrout = rgbb  + (size_t)P_ * 12;         // P x 40
  float* pe    = xrout + (size_t)P_ * 40;         // 128 x 8
  size_t need  = ((size_t)P_ * 228 + 1024) * sizeof(float);
  if (ws_size < need) return;

  float* outRad  = (float*)d_out;                 // P x 3
  float* outSrdf = outRad + (size_t)P_ * 3;       // P
  float* outPip  = outSrdf + (size_t)P_;          // B x NV x 2 x RN x SN

  hipFuncSetAttribute((const void*)ray_layer_kernel,
                      hipFuncAttributeMaxDynamicSharedMemorySize, RAY_SMEM);

  pe_kernel<<<1, 1024, 0, stream>>>(pe);
  geom_kernel<<<(B_ * NV_ * RN_ * SN_) / 256, 256, 0, stream>>>(
      point3D, simgs, poses, poses_inv, refpi, feat, vtok,
      xview, dirb, maskb, rgbb, outPip);
  view_layer_kernel<<<P_ / 16, 128, 0, stream>>>(
      xview, vW[0], vW[1], vW[2], vW[3], vW[4], vW[5], vW[6], vW[7], vW[8], vW[9]);
  ray_layer_kernel<<<B_ * RN_, 256, RAY_SMEM, stream>>>(
      xview, pe, rW[0], rW[1], rW[2], rW[3], rW[4], rW[5], rW[6], rW[7], rW[8], rW[9],
      xrout);
  density_kernel<<<P_ / 256, 256, 0, stream>>>(
      xrout, dW[0], dW[1], dW[2], dW[3], dW[4], dW[5], outSrdf);
  radiance_kernel<<<P_ / 256, 256, 0, stream>>>(
      xview, dirb, maskb, rgbb, cW[0], cW[1], cW[2], cW[3], cW[4], cW[5], outRad);
}
